// FreqMoEBlock_76201309766081
// MI455X (gfx1250) — matrix operand order repaired and checked by reading, not compiled, not run on hardware
//
#include <hip/hip_runtime.h>
#include <math.h>

#define E_    8
#define C_    128
#define L_    1024
#define B_    32
#define FB_   513            // L/2+1
#define HD_   1026
#define BC_   (B_ * C_)      // 4096
#define FBP_  544            // FB padded to multiple of 32 (>= 513+31)
#define HDP_  1088           // HD padded to multiple of 64 (17 * 64)
#define FBP2_ 576            // comb row padding: multiple of 64 >= 513

typedef __attribute__((ext_vector_type(16))) _Float16 v16h;
typedef __attribute__((ext_vector_type(8)))  float    v8f;
typedef __attribute__((ext_vector_type(4)))  _Float16 v4h;

// ---------------------------------------------------------------------------
// CDNA5 async global->LDS copy (ASYNCcnt-tracked DMA into LDS, no VGPR hop)
// ---------------------------------------------------------------------------
__device__ __forceinline__ unsigned lds_off_of(const void* p) {
    return (unsigned)(uintptr_t)p;   // low 32 bits of generic LDS ptr = LDS addr
}
__device__ __forceinline__ void async_copy_b128(unsigned lds_off, const void* gptr) {
    asm volatile("global_load_async_to_lds_b128 %0, %1, off"
                 :: "v"(lds_off), "v"((unsigned long long)(uintptr_t)gptr)
                 : "memory");
}
__device__ __forceinline__ void wait_async0() {
    asm volatile("s_wait_asynccnt 0x0" ::: "memory");
}

// ---------------------------------------------------------------------------
// In-LDS radix-2 1024-point FFT (256 threads). sign=-1: forward, +1: inverse.
// ---------------------------------------------------------------------------
__device__ __forceinline__ void fft1024(float* re, float* im, float sign) {
    const int tid = threadIdx.x;
    for (int i = tid; i < 1024; i += 256) {               // bit reversal (10b)
        int j = (int)(__brev((unsigned)i) >> 22);
        if (j > i) {
            float tr = re[i]; re[i] = re[j]; re[j] = tr;
            float ti = im[i]; im[i] = im[j]; im[j] = ti;
        }
    }
    __syncthreads();
    for (int s = 1; s <= 10; ++s) {
        const int half = 1 << (s - 1);
        const float ang = sign * 6.28318530717958647692f / (float)(1 << s);
        for (int t = tid; t < 512; t += 256) {
            const int k   = t & (half - 1);
            const int grp = t >> (s - 1);
            const int i0  = (grp << s) + k;
            const int i1  = i0 + half;
            float sw, cw;
            __sincosf(ang * (float)k, &sw, &cw);
            const float vr = re[i1], vi = im[i1];
            const float tr = vr * cw - vi * sw;
            const float ti = vr * sw + vi * cw;
            const float ur = re[i0], ui = im[i0];
            re[i0] = ur + tr; im[i0] = ui + ti;
            re[i1] = ur - tr; im[i1] = ui - ti;
        }
        __syncthreads();
    }
}

__global__ void zero_f32_kernel(float* p, int n) {
    int i = blockIdx.x * blockDim.x + threadIdx.x;
    if (i < n) p[i] = 0.0f;
}

// ---------------------------------------------------------------------------
// Forward rFFT per (b,c) row: f16 spectra (padded) + mean |Xf| accumulation
// ---------------------------------------------------------------------------
__global__ void __launch_bounds__(256) fft_fwd_kernel(
    const float* __restrict__ X,          // (BC, L)
    _Float16* __restrict__ xhr,           // (BC, FBP)
    _Float16* __restrict__ xhi,           // (BC, FBP)
    float* __restrict__ mag)              // (B, FB)  pre-zeroed
{
    __shared__ float re[1024];
    __shared__ float im[1024];
    const int row = blockIdx.x;           // b*C + c
    const int b   = row >> 7;
    const float* xp = X + (size_t)row * L_;
    async_copy_b128(lds_off_of(&re[threadIdx.x * 4]), xp + threadIdx.x * 4);
    for (int i = threadIdx.x; i < 1024; i += 256) im[i] = 0.0f;
    wait_async0();
    __syncthreads();
    fft1024(re, im, -1.0f);
    _Float16* pr = xhr + (size_t)row * FBP_;
    _Float16* pi = xhi + (size_t)row * FBP_;
    for (int f = threadIdx.x; f < FBP_; f += 256) {
        const float r  = (f < FB_) ? re[f] : 0.0f;
        const float ii = (f < FB_) ? im[f] : 0.0f;
        pr[f] = (_Float16)r;
        pi[f] = (_Float16)ii;
        if (f < FB_)
            atomicAdd(&mag[b * FB_ + f], sqrtf(r * r + ii * ii) * (1.0f / (float)C_));
    }
}

// ---------------------------------------------------------------------------
// Expert band boundaries: sigmoid -> sort([0, raw, 1]) -> idx -> starts/ends
// ---------------------------------------------------------------------------
__global__ void bounds_kernel(const float* __restrict__ bp, int* __restrict__ bands) {
    if (threadIdx.x == 0 && blockIdx.x == 0) {
        float v[9];
        v[0] = 0.0f; v[8] = 1.0f;
        for (int i = 0; i < 7; ++i) v[i + 1] = 1.0f / (1.0f + __expf(-bp[i]));
        for (int i = 1; i < 9; ++i) {
            float x = v[i]; int j = i - 1;
            while (j >= 0 && v[j] > x) { v[j + 1] = v[j]; --j; }
            v[j + 1] = x;
        }
        int idx[9];
        for (int i = 0; i < 9; ++i) idx[i] = (int)(v[i] * (float)(FB_ - 1));
        for (int e = 0; e < E_; ++e) {
            bands[e]     = idx[e];
            bands[8 + e] = (e < E_ - 1) ? idx[e + 1] : FB_;
        }
    }
}

// ---------------------------------------------------------------------------
// Gate: logits = mag @ gate_W.T + gate_b ; softmax over experts. (32x8 threads)
// ---------------------------------------------------------------------------
__global__ void __launch_bounds__(256) gate_kernel(
    const float* __restrict__ mag, const float* __restrict__ gW,
    const float* __restrict__ gb, float* __restrict__ gate)
{
    __shared__ float lg[B_][E_];
    const int t = threadIdx.x;
    const int b = t >> 3;
    const int e = t & 7;
    const float* mp = mag + (size_t)b * FB_;
    const float* wp = gW  + (size_t)e * FB_;
    float acc = gb[e];
    for (int f = 0; f < FB_; ++f) acc += mp[f] * wp[f];
    lg[b][e] = acc;
    __syncthreads();
    float mx = lg[b][0];
    for (int j = 1; j < E_; ++j) mx = fmaxf(mx, lg[b][j]);
    float sum = 0.0f;
    for (int j = 0; j < E_; ++j) sum += __expf(lg[b][j] - mx);
    gate[b * E_ + e] = __expf(acc - mx) / sum;
}

// ---------------------------------------------------------------------------
// WMMA fragment helpers (wave32 CDNA5 layouts, CDNA5 ISA §7.12.2)
// ---------------------------------------------------------------------------
__device__ __forceinline__ v16h frag_a(const _Float16* lds, int m_base, int lane) {
    const _Float16* p = lds + (size_t)(m_base + (lane & 15)) * 32 + ((lane >> 4) << 3);
    v16h a;
#pragma unroll
    for (int j = 0; j < 8; ++j) { a[j] = p[j]; a[8 + j] = p[16 + j]; }
    return a;
}
__device__ __forceinline__ v16h frag_b(const _Float16* lds, int n_base, int lane) {
    const _Float16* p = lds + (size_t)(n_base + (lane & 15)) * 32 + ((lane >> 4) << 3);
    v16h b;
#pragma unroll
    for (int j = 0; j < 8; ++j) { b[j] = p[j]; b[8 + j] = p[16 + j]; }
    return b;
}
__device__ __forceinline__ v8f wmma_f16(v16h a, v16h b, v8f c) {
    return __builtin_amdgcn_wmma_f32_16x16x32_f16(false, a, false, b, (short)0, c, false, false);
}

// ---------------------------------------------------------------------------
// B-tile staging: W (f32, 4B-aligned rows) -> f16 LDS [n][32], packed b64 store
// ---------------------------------------------------------------------------
__device__ __forceinline__ void stage_w_tile(
    const float* __restrict__ Wr, const float* __restrict__ Wi,
    _Float16* lb_r, _Float16* lb_i,
    int tid, size_t row_base, size_t row_stride,
    int n0, int nmax, int k0, int kmax)
{
#pragma unroll
    for (int i = 0; i < 2; ++i) {
        const int c  = tid + i * 256;   // 0..511 quads
        const int n  = c >> 3;          // 0..63
        const int k4 = (c & 7) * 4;
        const size_t gb = row_base + (size_t)n * row_stride + k4;
        const bool nok = (n0 + n) < nmax;
        v4h pr, pi;
#pragma unroll
        for (int j = 0; j < 4; ++j) {
            const bool ok = nok && ((k0 + k4 + j) < kmax);
            pr[j] = ok ? (_Float16)Wr[gb + j] : (_Float16)0.0f;
            pi[j] = ok ? (_Float16)Wi[gb + j] : (_Float16)0.0f;
        }
        *(v4h*)(&lb_r[n * 32 + k4]) = pr;
        *(v4h*)(&lb_i[n * 32 + k4]) = pi;
    }
}

// ---------------------------------------------------------------------------
// GEMM-1 (per expert): h = relu(W1 (complex) x + b1), K limited to band.
// Block 256 thr = 8 waves; tile 128x64; wave 32x32 (2x2 WMMA tiles); Kstep 32.
// ---------------------------------------------------------------------------
__global__ void __launch_bounds__(256) gemm1_kernel(
    const _Float16* __restrict__ xr, const _Float16* __restrict__ xi, // (BC, FBP)
    const float* __restrict__ W1r, const float* __restrict__ W1i,     // (E, HD, FB)
    const float* __restrict__ b1r, const float* __restrict__ b1i,     // (E, HD)
    const int* __restrict__ bands,                                    // starts[8], ends[8]
    _Float16* __restrict__ hro, _Float16* __restrict__ hio)           // (E, BC, HDP)
{
    const int e    = blockIdx.z;
    const int m0   = blockIdx.x * 128;
    const int h0   = blockIdx.y * 64;
    const int tid  = threadIdx.x;
    const int lane = tid & 31;
    const int wave = tid >> 5;
    const int wm   = (wave & 3) * 32;
    const int wn   = (wave >> 2) * 32;

    __shared__ __align__(16) unsigned char smem[24576];   // 24KB
    _Float16* la_r = (_Float16*)smem;                     // 128x32 = 8KB
    _Float16* la_i = la_r + 128 * 32;                     // 8KB
    _Float16* lb_r = la_i + 128 * 32;                     // 64x32 = 4KB
    _Float16* lb_i = lb_r + 64 * 32;                      // 4KB

    const int s  = bands[e];
    const int en = bands[8 + e];
    const int k0start = (s & ~31);

    // hoisted per-thread async chunk pointers (advance 32 halves / iter)
    const _Float16* paR[2];
    const _Float16* paI[2];
    unsigned ldsAr[2], ldsAi[2];
#pragma unroll
    for (int i = 0; i < 2; ++i) {
        const int c  = tid + i * 256;   // 0..511 b128 chunks
        const int m  = c >> 2;
        const int k8 = (c & 3) * 8;
        paR[i] = xr + (size_t)(m0 + m) * FBP_ + k0start + k8;
        paI[i] = xi + (size_t)(m0 + m) * FBP_ + k0start + k8;
        ldsAr[i] = lds_off_of(&la_r[m * 32 + k8]);
        ldsAi[i] = lds_off_of(&la_i[m * 32 + k8]);
    }

    const v8f vz = {0.f, 0.f, 0.f, 0.f, 0.f, 0.f, 0.f, 0.f};
    v8f acc_rr[2][2], acc_tt[2][2], acc_mi[2][2];
#pragma unroll
    for (int mt = 0; mt < 2; ++mt)
#pragma unroll
        for (int nt = 0; nt < 2; ++nt) {
            acc_rr[mt][nt] = vz; acc_tt[mt][nt] = vz; acc_mi[mt][nt] = vz;
        }

    for (int k0 = k0start; k0 < en; k0 += 32) {
        if ((k0 >= s) && (k0 + 32 <= en)) {   // interior tile: async DMA
#pragma unroll
            for (int i = 0; i < 2; ++i) {
                async_copy_b128(ldsAr[i], paR[i]);
                async_copy_b128(ldsAi[i], paI[i]);
            }
        } else {                               // boundary tile: band-masked
#pragma unroll
            for (int i = 0; i < 16; ++i) {
                const int idx = tid + i * 256;
                const int m = idx >> 5, k = idx & 31;
                const int gk = k0 + k;
                const bool inb = (gk >= s) && (gk < en);
                const size_t ga = (size_t)(m0 + m) * FBP_ + gk;
                la_r[idx] = inb ? xr[ga] : (_Float16)0.0f;
                la_i[idx] = inb ? xi[ga] : (_Float16)0.0f;
            }
        }
        stage_w_tile(W1r, W1i, lb_r, lb_i, tid,
                     ((size_t)e * HD_ + h0) * FB_ + k0, FB_,
                     h0, HD_, k0, FB_);
        wait_async0();
        __syncthreads();

        v16h ar[2], ai[2], br[2], bi[2];
#pragma unroll
        for (int mt = 0; mt < 2; ++mt) {
            ar[mt] = frag_a(la_r, wm + mt * 16, lane);
            ai[mt] = frag_a(la_i, wm + mt * 16, lane);
        }
#pragma unroll
        for (int nt = 0; nt < 2; ++nt) {
            br[nt] = frag_b(lb_r, wn + nt * 16, lane);
            bi[nt] = frag_b(lb_i, wn + nt * 16, lane);
        }
#pragma unroll
        for (int mt = 0; mt < 2; ++mt)
#pragma unroll
            for (int nt = 0; nt < 2; ++nt) {
                acc_rr[mt][nt] = wmma_f16(ar[mt], br[nt], acc_rr[mt][nt]); // xr*W1r
                acc_tt[mt][nt] = wmma_f16(ai[mt], bi[nt], acc_tt[mt][nt]); // xi*W1i
                acc_mi[mt][nt] = wmma_f16(ar[mt], bi[nt], acc_mi[mt][nt]); // xr*W1i
                acc_mi[mt][nt] = wmma_f16(ai[mt], br[nt], acc_mi[mt][nt]); // + xi*W1r
            }
        __syncthreads();
#pragma unroll
        for (int i = 0; i < 2; ++i) { paR[i] += 32; paI[i] += 32; }
    }

    // ---- epilogue: bias+relu -> LDS tile -> cooperative b128 stores ----
    _Float16* ht = (_Float16*)smem;           // 128x64 f16 = 16KB (reuses staging)
    const int lcol = wn + (lane & 15);        // column pieces per nt below
    // pass 1: hr
    __syncthreads();
#pragma unroll
    for (int mt = 0; mt < 2; ++mt)
#pragma unroll
        for (int nt = 0; nt < 2; ++nt)
#pragma unroll
            for (int r = 0; r < 8; ++r) {
                const int lr = wm + mt * 16 + r + ((lane >> 4) << 3);
                const int lc = lcol + nt * 16;
                const int gh = h0 + lc;
                const float bv = (gh < HD_) ? b1r[e * HD_ + gh] : 0.0f;
                const float vr = acc_rr[mt][nt][r] - acc_tt[mt][nt][r] + bv;
                ht[lr * 64 + lc] = (gh < HD_) ? (_Float16)fmaxf(vr, 0.0f) : (_Float16)0.0f;
            }
    __syncthreads();
#pragma unroll
    for (int i = 0; i < 4; ++i) {
        const int c  = tid + i * 256;   // 0..1023 chunks of 8 halves
        const int m  = c >> 3;
        const int k8 = (c & 7) * 8;
        *(uint4*)(hro + ((size_t)e * BC_ + m0 + m) * HDP_ + h0 + k8) =
            *(const uint4*)(ht + m * 64 + k8);
    }
    __syncthreads();
    // pass 2: hi
#pragma unroll
    for (int mt = 0; mt < 2; ++mt)
#pragma unroll
        for (int nt = 0; nt < 2; ++nt)
#pragma unroll
            for (int r = 0; r < 8; ++r) {
                const int lr = wm + mt * 16 + r + ((lane >> 4) << 3);
                const int lc = lcol + nt * 16;
                const int gh = h0 + lc;
                const float bv = (gh < HD_) ? b1i[e * HD_ + gh] : 0.0f;
                const float vi = acc_mi[mt][nt][r] + bv;
                ht[lr * 64 + lc] = (gh < HD_) ? (_Float16)fmaxf(vi, 0.0f) : (_Float16)0.0f;
            }
    __syncthreads();
#pragma unroll
    for (int i = 0; i < 4; ++i) {
        const int c  = tid + i * 256;
        const int m  = c >> 3;
        const int k8 = (c & 7) * 8;
        *(uint4*)(hio + ((size_t)e * BC_ + m0 + m) * HDP_ + h0 + k8) =
            *(const uint4*)(ht + m * 64 + k8);
    }
}

// ---------------------------------------------------------------------------
// GEMM-2 + gate-weighted expert combine:
//   comb = sum_e gate[b,e] * (W2[e] (complex) h[e] + b2[e])
// comb rows padded to FBP2 so b128 stores never cross rows (pad cols == 0).
// ---------------------------------------------------------------------------
__global__ void __launch_bounds__(256) gemm2_kernel(
    const _Float16* __restrict__ hr, const _Float16* __restrict__ hi, // (E, BC, HDP)
    const float* __restrict__ W2r, const float* __restrict__ W2i,     // (E, FB, HD)
    const float* __restrict__ b2r, const float* __restrict__ b2i,     // (E, FB)
    const float* __restrict__ gate,                                   // (B, E)
    float* __restrict__ comb_r, float* __restrict__ comb_i)           // (BC, FBP2)
{
    const int m0   = blockIdx.x * 128;
    const int f0   = blockIdx.y * 64;
    const int bb   = m0 >> 7;
    const int tid  = threadIdx.x;
    const int lane = tid & 31;
    const int wave = tid >> 5;
    const int wm   = (wave & 3) * 32;
    const int wn   = (wave >> 2) * 32;

    __shared__ __align__(16) unsigned char smem[32768];   // 32KB
    _Float16* la_r = (_Float16*)smem;
    _Float16* la_i = la_r + 128 * 32;
    _Float16* lb_r = la_i + 128 * 32;
    _Float16* lb_i = lb_r + 64 * 32;

    // LDS destinations for async A chunks (constant across e and k)
    unsigned ldsAr[2], ldsAi[2];
    int chM[2], chK8[2];
#pragma unroll
    for (int i = 0; i < 2; ++i) {
        const int c  = tid + i * 256;
        chM[i]  = c >> 2;
        chK8[i] = (c & 3) * 8;
        ldsAr[i] = lds_off_of(&la_r[chM[i] * 32 + chK8[i]]);
        ldsAi[i] = lds_off_of(&la_i[chM[i] * 32 + chK8[i]]);
    }

    const v8f vz = {0.f, 0.f, 0.f, 0.f, 0.f, 0.f, 0.f, 0.f};
    v8f cr[2][2], ci[2][2];
#pragma unroll
    for (int mt = 0; mt < 2; ++mt)
#pragma unroll
        for (int nt = 0; nt < 2; ++nt) { cr[mt][nt] = vz; ci[mt][nt] = vz; }

    for (int e = 0; e < E_; ++e) {
        const float g = gate[bb * E_ + e];
        v8f t_rr[2][2], t_ii[2][2], t_mi[2][2];
#pragma unroll
        for (int mt = 0; mt < 2; ++mt)
#pragma unroll
            for (int nt = 0; nt < 2; ++nt) {
                t_rr[mt][nt] = vz; t_ii[mt][nt] = vz; t_mi[mt][nt] = vz;
            }

        const _Float16* pHr[2];
        const _Float16* pHi[2];
#pragma unroll
        for (int i = 0; i < 2; ++i) {
            pHr[i] = hr + ((size_t)e * BC_ + m0 + chM[i]) * HDP_ + chK8[i];
            pHi[i] = hi + ((size_t)e * BC_ + m0 + chM[i]) * HDP_ + chK8[i];
        }

        for (int k0 = 0; k0 < HDP_; k0 += 32) {
#pragma unroll
            for (int i = 0; i < 2; ++i) {        // pure async DMA, no masking
                async_copy_b128(ldsAr[i], pHr[i]);
                async_copy_b128(ldsAi[i], pHi[i]);
            }
            stage_w_tile(W2r, W2i, lb_r, lb_i, tid,
                         ((size_t)e * FB_ + f0) * HD_ + k0, HD_,
                         f0, FB_, k0, HD_);
            wait_async0();
            __syncthreads();

            v16h ar[2], ai[2], br[2], bi[2];
#pragma unroll
            for (int mt = 0; mt < 2; ++mt) {
                ar[mt] = frag_a(la_r, wm + mt * 16, lane);
                ai[mt] = frag_a(la_i, wm + mt * 16, lane);
            }
#pragma unroll
            for (int nt = 0; nt < 2; ++nt) {
                br[nt] = frag_b(lb_r, wn + nt * 16, lane);
                bi[nt] = frag_b(lb_i, wn + nt * 16, lane);
            }
#pragma unroll
            for (int mt = 0; mt < 2; ++mt)
#pragma unroll
                for (int nt = 0; nt < 2; ++nt) {
                    t_rr[mt][nt] = wmma_f16(ar[mt], br[nt], t_rr[mt][nt]); // hr*W2r
                    t_ii[mt][nt] = wmma_f16(ai[mt], bi[nt], t_ii[mt][nt]); // hi*W2i
                    t_mi[mt][nt] = wmma_f16(ar[mt], bi[nt], t_mi[mt][nt]); // hr*W2i
                    t_mi[mt][nt] = wmma_f16(ai[mt], br[nt], t_mi[mt][nt]); // + hi*W2r
                }
            __syncthreads();
#pragma unroll
            for (int i = 0; i < 2; ++i) { pHr[i] += 32; pHi[i] += 32; }
        }

        // fold: comb += g * (y + b2)   (pad cols: t==0 and bias==0 -> stays 0)
#pragma unroll
        for (int mt = 0; mt < 2; ++mt)
#pragma unroll
            for (int nt = 0; nt < 2; ++nt)
#pragma unroll
                for (int r = 0; r < 8; ++r) {
                    const int gf = f0 + wn + nt * 16 + (lane & 15);
                    const float brv = (gf < FB_) ? b2r[e * FB_ + gf] : 0.0f;
                    const float biv = (gf < FB_) ? b2i[e * FB_ + gf] : 0.0f;
                    cr[mt][nt][r] += g * (t_rr[mt][nt][r] - t_ii[mt][nt][r] + brv);
                    ci[mt][nt][r] += g * (t_mi[mt][nt][r] + biv);
                }
    }

    // ---- epilogue: LDS repack -> cooperative b128 stores (rows padded) ----
    float* ft = (float*)smem;   // 128x64 f32 = 32KB (reuses staging)
    // comb_r
    __syncthreads();
#pragma unroll
    for (int mt = 0; mt < 2; ++mt)
#pragma unroll
        for (int nt = 0; nt < 2; ++nt)
#pragma unroll
            for (int r = 0; r < 8; ++r) {
                const int lr = wm + mt * 16 + r + ((lane >> 4) << 3);
                const int lc = wn + nt * 16 + (lane & 15);
                ft[lr * 64 + lc] = cr[mt][nt][r];
            }
    __syncthreads();
#pragma unroll
    for (int i = 0; i < 8; ++i) {
        const int c  = tid + i * 256;   // 0..2047 float4 chunks
        const int m  = c >> 4;
        const int k4 = (c & 15) * 4;
        *(float4*)(comb_r + (size_t)(m0 + m) * FBP2_ + f0 + k4) =
            *(const float4*)(ft + m * 64 + k4);
    }
    __syncthreads();
    // comb_i
#pragma unroll
    for (int mt = 0; mt < 2; ++mt)
#pragma unroll
        for (int nt = 0; nt < 2; ++nt)
#pragma unroll
            for (int r = 0; r < 8; ++r) {
                const int lr = wm + mt * 16 + r + ((lane >> 4) << 3);
                const int lc = wn + nt * 16 + (lane & 15);
                ft[lr * 64 + lc] = ci[mt][nt][r];
            }
    __syncthreads();
#pragma unroll
    for (int i = 0; i < 8; ++i) {
        const int c  = tid + i * 256;
        const int m  = c >> 4;
        const int k4 = (c & 15) * 4;
        *(float4*)(comb_i + (size_t)(m0 + m) * FBP2_ + f0 + k4) =
            *(const float4*)(ft + m * 64 + k4);
    }
}

// ---------------------------------------------------------------------------
// irFFT: Hermitian-extend comb spectrum, inverse FFT in LDS, write real signal
// ---------------------------------------------------------------------------
__global__ void __launch_bounds__(256) ifft_kernel(
    const float* __restrict__ comb_r, const float* __restrict__ comb_i, // (BC, FBP2)
    float* __restrict__ out)                                            // (BC, L)
{
    __shared__ float re[1024];
    __shared__ float im[1024];
    const int row = blockIdx.x;
    const float* cr = comb_r + (size_t)row * FBP2_;
    const float* ci = comb_i + (size_t)row * FBP2_;
    for (int i = threadIdx.x; i < 1024; i += 256) {
        if (i <= 512) { re[i] = cr[i];        im[i] = ci[i]; }
        else          { re[i] = cr[1024 - i]; im[i] = -ci[1024 - i]; }
    }
    __syncthreads();
    fft1024(re, im, +1.0f);
    float* op = out + (size_t)row * L_;
    for (int i = threadIdx.x; i < 1024; i += 256)
        op[i] = re[i] * (1.0f / 1024.0f);
}

// ---------------------------------------------------------------------------
// Host launcher
// ---------------------------------------------------------------------------
extern "C" void kernel_launch(void* const* d_in, const int* in_sizes, int n_in,
                              void* d_out, int out_size, void* d_ws, size_t ws_size,
                              hipStream_t stream) {
    (void)in_sizes; (void)n_in; (void)out_size; (void)ws_size;

    const float* X   = (const float*)d_in[0];
    const float* bp  = (const float*)d_in[1];
    const float* W1r = (const float*)d_in[2];
    const float* W1i = (const float*)d_in[3];
    const float* b1r = (const float*)d_in[4];
    const float* b1i = (const float*)d_in[5];
    const float* W2r = (const float*)d_in[6];
    const float* W2i = (const float*)d_in[7];
    const float* b2r = (const float*)d_in[8];
    const float* b2i = (const float*)d_in[9];
    const float* gW  = (const float*)d_in[10];
    const float* gb  = (const float*)d_in[11];
    float* out = (float*)d_out;

    char* w = (char*)d_ws;
    size_t off = 0;
    auto alloc = [&](size_t bytes) -> void* {
        void* p = (void*)(w + off);
        off = (off + bytes + 255) & ~(size_t)255;
        return p;
    };

    _Float16* xhr   = (_Float16*)alloc((size_t)BC_ * FBP_ * sizeof(_Float16));
    _Float16* xhi   = (_Float16*)alloc((size_t)BC_ * FBP_ * sizeof(_Float16));
    float*    mag   = (float*)   alloc((size_t)B_ * FB_ * sizeof(float));
    float*    gate  = (float*)   alloc((size_t)B_ * E_ * sizeof(float));
    int*      bands = (int*)     alloc(16 * sizeof(int));
    _Float16* hr    = (_Float16*)alloc((size_t)E_ * BC_ * HDP_ * sizeof(_Float16));
    _Float16* hi    = (_Float16*)alloc((size_t)E_ * BC_ * HDP_ * sizeof(_Float16));
    float*    cbr   = (float*)   alloc((size_t)BC_ * FBP2_ * sizeof(float));
    float*    cbi   = (float*)   alloc((size_t)BC_ * FBP2_ * sizeof(float));

    {
        const int n = B_ * FB_;
        zero_f32_kernel<<<(n + 255) / 256, 256, 0, stream>>>(mag, n);
    }
    fft_fwd_kernel<<<BC_, 256, 0, stream>>>(X, xhr, xhi, mag);
    bounds_kernel<<<1, 32, 0, stream>>>(bp, bands);
    gate_kernel<<<1, 256, 0, stream>>>(mag, gW, gb, gate);
    gemm1_kernel<<<dim3(BC_ / 128, HDP_ / 64, E_), 256, 0, stream>>>(
        xhr, xhi, W1r, W1i, b1r, b1i, bands, hr, hi);
    gemm2_kernel<<<dim3(BC_ / 128, (FB_ + 63) / 64, 1), 256, 0, stream>>>(
        hr, hi, W2r, W2i, b2r, b2i, gate, cbr, cbi);
    ifft_kernel<<<BC_, 256, 0, stream>>>(cbr, cbi, out);
}
